// selfAttention_77309412040
// MI455X (gfx1250) — matrix operand order repaired and checked by reading, not compiled, not run on hardware
//
#include <hip/hip_runtime.h>
#include <hip/hip_bf16.h>

// ---------------------------------------------------------------------------
// Self-attention for MI455X (gfx1250): bf16 WMMA everywhere, flash-attention
// streaming softmax, wave32 (4 waves / 128-thread block, one 16-row tile/wave).
// All WMMA operands are fed by wide K-contiguous loads (no scalar gathers):
//  - weights stay row-major (already B-fragment-friendly), just cast to bf16
//  - V is stored transposed per head so P@V fragments are contiguous too.
// ---------------------------------------------------------------------------

#define HEADS 16
#define HD    64
#define EMBED 1024
#define BATCH 2
#define SEQ   2048

typedef __bf16 bf16;
typedef __attribute__((ext_vector_type(16))) __bf16 v16bf;
typedef __attribute__((ext_vector_type(8)))  __bf16 v8bf;
typedef __attribute__((ext_vector_type(8)))  float  v8f;

// D = A(16x32 bf16) * B(32x16 bf16) + C(16x16 f32)
__device__ __forceinline__ v8f wmma_bf16(v16bf a, v16bf b, v8f c) {
  return __builtin_amdgcn_wmma_f32_16x16x32_bf16(
      /*neg_a=*/false, a, /*neg_b=*/false, b,
      /*c_mod=*/(short)0, c, /*reuse_a=*/false, /*reuse_b=*/false);
}

// ---- fragment loaders (wave32; A/B share contiguous K order k = 16*(l/16) + s) ----

// A 16x32 bf16: lane l holds row m=l&15; lanes 0-15 K={0..15},
// lanes 16-31 K={16..31}.  p = tile origin, ld = row stride (elements).
__device__ __forceinline__ v16bf load_a_bf16(const bf16* p, int ld) {
  int l  = threadIdx.x & 31;
  int m  = l & 15;
  int hs = (l >> 4) << 4;               // 0 or 16
  const bf16* r = p + (size_t)m * ld + hs;
  v8bf lo = *(const v8bf*)(r);          // K = hs .. hs+7
  v8bf hi = *(const v8bf*)(r + 8);      // K = hs+8 .. hs+15
  v16bf out;
#pragma unroll
  for (int i = 0; i < 8; ++i) { out[i] = lo[i]; out[i + 8] = hi[i]; }
  return out;
}

// A 16x32 from f32 source (two 32B vector loads), converting to bf16.
__device__ __forceinline__ v16bf load_a_f32(const float* p, int ld) {
  int l  = threadIdx.x & 31;
  int m  = l & 15;
  int hs = (l >> 4) << 4;
  const float* r = p + (size_t)m * ld + hs;
  v8f lo = *(const v8f*)(r);
  v8f hi = *(const v8f*)(r + 8);
  v16bf out;
#pragma unroll
  for (int i = 0; i < 8; ++i) {
    out[i]     = (bf16)lo[i];
    out[i + 8] = (bf16)hi[i];
  }
  return out;
}

// B 32x16 bf16 where element (k,n) lives at p[n*ld + k]: one aligned v16bf
// load per lane.  Covers row-major weights (W[e][d], k=d, n=e), row-major K
// (K[key][d], k=d, n=key) and transposed V (Vt[d][key], k=key, n=d).
__device__ __forceinline__ v16bf load_b_kcontig(const bf16* p, size_t ld) {
  int l  = threadIdx.x & 31;
  int n  = l & 15;
  int kb = (l >> 4) << 4;               // 0 or 16
  return *(const v16bf*)(p + (size_t)n * ld + kb);
}

// C/D 16x16 f32: lane l holds col n=l&15, rows rb..rb+7 with rb=(l>>4)*8.
__device__ __forceinline__ void store_c_bf16(bf16* p, int ld, v8f c) {
  int l  = threadIdx.x & 31;
  int n  = l & 15;
  int rb = (l >> 4) << 3;
#pragma unroll
  for (int r = 0; r < 8; ++r) p[(size_t)(rb + r) * ld + n] = (bf16)c[r];
}

// C/D store transposed: element (m,n) -> p[n*ld + m].  Each lane's 8 rows are
// contiguous in memory -> single packed 16-byte store.
__device__ __forceinline__ void store_c_bf16_T(bf16* p, size_t ld, v8f c) {
  int l  = threadIdx.x & 31;
  int n  = l & 15;
  int rb = (l >> 4) << 3;
  v8bf pk;
#pragma unroll
  for (int r = 0; r < 8; ++r) pk[r] = (bf16)c[r];
  *(v8bf*)(p + (size_t)n * ld + rb) = pk;
}

// ---------------------------------------------------------------------------
// Kernel 0: cast weights to bf16.  NO transpose: row-major W[e][d] is already
// K-contiguous for B fragments.
// ---------------------------------------------------------------------------
__global__ void __launch_bounds__(256)
convert_weights(const float* __restrict__ Wq, const float* __restrict__ Wk,
                const float* __restrict__ Wv, const float* __restrict__ Wo,
                bf16* __restrict__ Wqb, bf16* __restrict__ Wkb,
                bf16* __restrict__ Wvb, bf16* __restrict__ Wob) {
  int tid = blockIdx.x * blockDim.x + threadIdx.x;
  if (tid < HD * HD) {
    Wqb[tid] = (bf16)Wq[tid];
    Wkb[tid] = (bf16)Wk[tid];
    Wvb[tid] = (bf16)Wv[tid];
  }
  for (int i = tid; i < EMBED * EMBED; i += gridDim.x * blockDim.x)
    Wob[i] = (bf16)Wo[i];
}

// ---------------------------------------------------------------------------
// Kernel 1: per-head QKV projection.  One wave = one 16-row tile of one
// (b,h) block.  Reference quirk: plain reshape -> x[b,h,l,d] sits at flat
// index ((b*H+h)*S+l)*64+d in the original [B,S,E] buffer.
// Q,K stored [bh][l][d]; V stored transposed [bh][d][key] for P@V fragments.
// ---------------------------------------------------------------------------
__global__ void __launch_bounds__(128)
proj_qkv(const float* __restrict__ q, const float* __restrict__ k,
         const float* __restrict__ v,
         const bf16* __restrict__ Wqb, const bf16* __restrict__ Wkb,
         const bf16* __restrict__ Wvb,
         bf16* __restrict__ Qbf, bf16* __restrict__ Kbf,
         bf16* __restrict__ Vt) {
  int warp = threadIdx.x >> 5;
  int tile = blockIdx.x * 4 + warp;             // 0 .. B*H*S/16-1
  int qtiles = SEQ / 16;
  int bh = tile / qtiles;
  int st = tile % qtiles;
  size_t base = (size_t)tile * 16 * HD;

  // ---- Q ----
  {
    v16bf a0 = load_a_f32(q + base, HD);        // K(d) = 0..31
    v16bf a1 = load_a_f32(q + base + 32, HD);   // K(d) = 32..63
#pragma unroll
    for (int nt = 0; nt < 4; ++nt) {
      v8f acc = {};
      acc = wmma_bf16(a0, load_b_kcontig(Wqb + nt * 16 * HD, HD), acc);
      acc = wmma_bf16(a1, load_b_kcontig(Wqb + nt * 16 * HD + 32, HD), acc);
      store_c_bf16(Qbf + base + nt * 16, HD, acc);
    }
  }
  // ---- K ----
  {
    v16bf a0 = load_a_f32(k + base, HD);
    v16bf a1 = load_a_f32(k + base + 32, HD);
#pragma unroll
    for (int nt = 0; nt < 4; ++nt) {
      v8f acc = {};
      acc = wmma_bf16(a0, load_b_kcontig(Wkb + nt * 16 * HD, HD), acc);
      acc = wmma_bf16(a1, load_b_kcontig(Wkb + nt * 16 * HD + 32, HD), acc);
      store_c_bf16(Kbf + base + nt * 16, HD, acc);
    }
  }
  // ---- V (transposed store: Vt[(bh*64 + d)*SEQ + key]) ----
  {
    v16bf a0 = load_a_f32(v + base, HD);
    v16bf a1 = load_a_f32(v + base + 32, HD);
#pragma unroll
    for (int nt = 0; nt < 4; ++nt) {
      v8f acc = {};
      acc = wmma_bf16(a0, load_b_kcontig(Wvb + nt * 16 * HD, HD), acc);
      acc = wmma_bf16(a1, load_b_kcontig(Wvb + nt * 16 * HD + 32, HD), acc);
      bf16* vtile = Vt + ((size_t)bh * HD + nt * 16) * SEQ + (size_t)st * 16;
      store_c_bf16_T(vtile, SEQ, acc);
    }
  }
}

// ---------------------------------------------------------------------------
// Kernel 2: flash attention.  One wave = 16 query rows of one (b,h).
// Streams keys/values in 32-wide chunks; online softmax with lane-aligned row
// statistics; P restaged through LDS (C-layout -> A-layout) as exp'd bf16.
// ---------------------------------------------------------------------------
__global__ void __launch_bounds__(128)
attn(const bf16* __restrict__ Qbf, const bf16* __restrict__ Kbf,
     const bf16* __restrict__ Vt, const int* __restrict__ mask,
     bf16* __restrict__ Obf) {
  __shared__ __align__(32) bf16 Pst[4][16 * 32];

  int warp = threadIdx.x >> 5;
  int lane = threadIdx.x & 31;
  int n    = lane & 15;
  int rb   = (lane >> 4) << 3;

  int qtiles = SEQ / 16;
  int tile   = blockIdx.x * 4 + warp;           // 0 .. B*H*S/16-1
  int bh     = tile / qtiles;
  int qt     = tile % qtiles;
  int b      = bh / HEADS;

  const bf16* qbase = Qbf + ((size_t)bh * SEQ + (size_t)qt * 16) * HD;
  const bf16* kbh   = Kbf + (size_t)bh * SEQ * HD;
  const bf16* vbh   = Vt + (size_t)bh * HD * SEQ;
  const int*  mrow0 = mask + (size_t)b * SEQ * SEQ + (size_t)(qt * 16) * SEQ;

  v16bf aq0 = load_a_bf16(qbase, HD);
  v16bf aq1 = load_a_bf16(qbase + 32, HD);

  v8f o0 = {}, o1 = {}, o2 = {}, o3 = {};
  float mrun[8], lrun[8];
#pragma unroll
  for (int r = 0; r < 8; ++r) { mrun[r] = -1.0e30f; lrun[r] = 0.0f; }

  const float inv_scale = 1.0f / 32.0f;         // 1/sqrt(EMBED), faithful quirk

  for (int kt = 0; kt < SEQ / 32; ++kt) {
    const bf16* kp = kbh + (size_t)kt * 32 * HD;
    __builtin_prefetch(kp + 32 * HD, 0, 1);              // next K chunk -> L2
    __builtin_prefetch(vbh + (size_t)(kt + 1) * 32, 0, 1); // next V columns

    // ---- scores: 16 queries x 32 keys, two 16x16 C tiles -----------------
    float x[2][8];
#pragma unroll
    for (int j = 0; j < 2; ++j) {
      v8f sc = {};
      sc = wmma_bf16(aq0, load_b_kcontig(kp + j * 16 * HD, HD), sc);
      sc = wmma_bf16(aq1, load_b_kcontig(kp + j * 16 * HD + 32, HD), sc);
      int key = kt * 32 + j * 16 + n;
#pragma unroll
      for (int r = 0; r < 8; ++r) {
        int mv = mrow0[(size_t)(rb + r) * SEQ + key];    // coalesced over lanes
        x[j][r] = mv ? sc[r] * inv_scale : -3.125e18f;   // (-1e20)/32
      }
    }

    // ---- online softmax: row stats align lane-wise with C layout ---------
#pragma unroll
    for (int r = 0; r < 8; ++r) {
      float cm = fmaxf(x[0][r], x[1][r]);
      cm = fmaxf(cm, __shfl_xor(cm, 1, 32));
      cm = fmaxf(cm, __shfl_xor(cm, 2, 32));
      cm = fmaxf(cm, __shfl_xor(cm, 4, 32));
      cm = fmaxf(cm, __shfl_xor(cm, 8, 32));    // row max within 16-lane group
      float nm   = fmaxf(mrun[r], cm);
      float corr = __expf(mrun[r] - nm);
      float p0   = __expf(x[0][r] - nm);
      float p1   = __expf(x[1][r] - nm);
      float cs   = p0 + p1;
      cs += __shfl_xor(cs, 1, 32);
      cs += __shfl_xor(cs, 2, 32);
      cs += __shfl_xor(cs, 4, 32);
      cs += __shfl_xor(cs, 8, 32);
      lrun[r] = lrun[r] * corr + cs;
      mrun[r] = nm;
      x[0][r] = p0; x[1][r] = p1;
      o0[r] *= corr; o1[r] *= corr; o2[r] *= corr; o3[r] *= corr;
    }

    // ---- restage P (C layout -> A layout) via LDS, as bf16 ---------------
    bf16* ps = &Pst[warp][0];
#pragma unroll
    for (int r = 0; r < 8; ++r) {
      ps[(rb + r) * 32 + n]      = (bf16)x[0][r];
      ps[(rb + r) * 32 + 16 + n] = (bf16)x[1][r];
    }
    __syncthreads();
    v16bf pa = load_a_bf16(ps, 32);             // P: 16x32 bf16

    // ---- O += P @ V_chunk: Vt is [d][key] so fragments are K-contiguous --
    const bf16* vp = vbh + (size_t)kt * 32;     // column block of Vt
    o0 = wmma_bf16(pa, load_b_kcontig(vp + (size_t)0  * SEQ, SEQ), o0);
    o1 = wmma_bf16(pa, load_b_kcontig(vp + (size_t)16 * SEQ, SEQ), o1);
    o2 = wmma_bf16(pa, load_b_kcontig(vp + (size_t)32 * SEQ, SEQ), o2);
    o3 = wmma_bf16(pa, load_b_kcontig(vp + (size_t)48 * SEQ, SEQ), o3);
    __syncthreads();
  }

  // ---- normalize by row sums and emit bf16 attention output --------------
#pragma unroll
  for (int r = 0; r < 8; ++r) {
    float inv = 1.0f / lrun[r];
    o0[r] *= inv; o1[r] *= inv; o2[r] *= inv; o3[r] *= inv;
  }
  bf16* obase = Obf + ((size_t)bh * SEQ + (size_t)qt * 16) * HD;
  store_c_bf16(obase + 0,  HD, o0);
  store_c_bf16(obase + 16, HD, o1);
  store_c_bf16(obase + 32, HD, o2);
  store_c_bf16(obase + 48, HD, o3);
}

// ---------------------------------------------------------------------------
// Kernel 3: output projection Y = X @ Wo^T + bo.  X (attention output in
// BHLD-flat order) IS the reference's final reshape layout -> treat directly
// as [B*S, 1024] row-major.  One wave = one 16x16 tile, K streamed in 32s.
// ---------------------------------------------------------------------------
__global__ void __launch_bounds__(128)
out_proj(const bf16* __restrict__ X, const bf16* __restrict__ Wob,
         const float* __restrict__ bo, float* __restrict__ out) {
  int warp = threadIdx.x >> 5;
  int lane = threadIdx.x & 31;
  int tile = blockIdx.x * 4 + warp;             // 0 .. (B*S/16)*(E/16)-1
  int ctiles = EMBED / 16;
  int rt = tile / ctiles;
  int ct = tile % ctiles;

  const bf16* xrow  = X + (size_t)rt * 16 * EMBED;
  const bf16* wbase = Wob + (size_t)(ct * 16) * EMBED;   // rows e=ct*16..+15
  v8f acc = {};
#pragma unroll 4
  for (int kc = 0; kc < EMBED / 32; ++kc) {
    if (kc + 1 < EMBED / 32) __builtin_prefetch(xrow + (kc + 1) * 32, 0, 1);
    v16bf a = load_a_bf16(xrow + kc * 32, EMBED);
    v16bf w = load_b_kcontig(wbase + kc * 32, EMBED);    // one v16bf per lane
    acc = wmma_bf16(a, w, acc);
  }

  int   n    = lane & 15;
  int   rb   = (lane >> 4) << 3;
  float bias = bo[ct * 16 + n];
#pragma unroll
  for (int r = 0; r < 8; ++r)
    out[((size_t)rt * 16 + rb + r) * EMBED + ct * 16 + n] = acc[r] + bias;
}

// ---------------------------------------------------------------------------
// Launch
// ---------------------------------------------------------------------------
extern "C" void kernel_launch(void* const* d_in, const int* in_sizes, int n_in,
                              void* d_out, int out_size, void* d_ws, size_t ws_size,
                              hipStream_t stream) {
  const float* value = (const float*)d_in[0];
  const float* key   = (const float*)d_in[1];
  const float* query = (const float*)d_in[2];
  const int*   mask  = (const int*)d_in[3];
  const float* Wq    = (const float*)d_in[4];
  const float* Wk    = (const float*)d_in[5];
  const float* Wv    = (const float*)d_in[6];
  const float* Wo    = (const float*)d_in[7];
  const float* bo    = (const float*)d_in[8];
  float*       out   = (float*)d_out;

  // workspace layout (bf16 regions, all 32B+ aligned)
  char* ws = (char*)d_ws;
  const size_t szW  = (size_t)HD * HD * sizeof(bf16);             //   8 KB each
  const size_t szWo = (size_t)EMBED * EMBED * sizeof(bf16);       //   2 MB
  const size_t szT  = (size_t)BATCH * SEQ * EMBED * sizeof(bf16); //   8 MB each
  bf16* Wqb = (bf16*)(ws);
  bf16* Wkb = (bf16*)(ws + szW);
  bf16* Wvb = (bf16*)(ws + 2 * szW);
  bf16* Wob = (bf16*)(ws + 3 * szW);
  bf16* Qbf = (bf16*)(ws + 3 * szW + szWo);
  bf16* Kbf = (bf16*)(ws + 3 * szW + szWo + szT);
  bf16* Vt  = (bf16*)(ws + 3 * szW + szWo + 2 * szT);
  bf16* Obf = (bf16*)(ws + 3 * szW + szWo + 3 * szT);

  convert_weights<<<256, 256, 0, stream>>>(Wq, Wk, Wv, Wo, Wqb, Wkb, Wvb, Wob);

  int tiles = BATCH * HEADS * SEQ / 16;          // 4096 (one per wave)
  proj_qkv<<<tiles / 4, 128, 0, stream>>>(query, key, value,
                                          Wqb, Wkb, Wvb, Qbf, Kbf, Vt);

  attn<<<tiles / 4, 128, 0, stream>>>(Qbf, Kbf, Vt, mask, Obf);

  int otiles = (BATCH * SEQ / 16) * (EMBED / 16); // 16384
  out_proj<<<otiles / 4, 128, 0, stream>>>(Obf, Wob, bo, out);
}
